// Encoder_7267084664867
// MI455X (gfx1250) — compile-verified, matrix operand order checked by reading, not run on hardware
//
#include <hip/hip_runtime.h>
#include <math.h>

typedef _Float16 half_t;
typedef __attribute__((ext_vector_type(16))) _Float16 v16h;
typedef __attribute__((ext_vector_type(8)))  _Float16 v8h;
typedef __attribute__((ext_vector_type(4)))  _Float16 v4h;
typedef __attribute__((ext_vector_type(8)))  float    v8f;
typedef __attribute__((ext_vector_type(4)))  float    v4f;

#define BATCH 4096
#define SEQ   128
#define FEAT  256

// ---------------------------------------------------------------------------
// Kernel 1: convert W1 / W2 (f32, 256x256 row-major) to f16 in workspace.
// ---------------------------------------------------------------------------
__global__ __launch_bounds__(256) void cvt_w_kernel(const float* __restrict__ W1,
                                                    const float* __restrict__ W2,
                                                    half_t* __restrict__ W1h,
                                                    half_t* __restrict__ W2h) {
    int i = blockIdx.x * blockDim.x + threadIdx.x;
    if (i < FEAT * FEAT) {
        W1h[i] = (half_t)W1[i];
        W2h[i] = (half_t)W2[i];
    }
}

// ---------------------------------------------------------------------------
// Kernel 2: masked sum pooling, streaming 512 MiB with NT b128 loads.
// One block per batch row; 64 lanes, each owns one float4 feature group.
// Output written as f16 (layer-1 A operand precision).
// ---------------------------------------------------------------------------
__global__ __launch_bounds__(64) void pool_kernel(const float* __restrict__ te,
                                                  const unsigned char* __restrict__ mask,
                                                  half_t* __restrict__ xh) {
    const int b = blockIdx.x;
    const int g = threadIdx.x;                 // 0..63 -> float4 group
    const v4f* row = (const v4f*)(te + (size_t)b * SEQ * FEAT);
    const unsigned char* m = mask + (size_t)b * SEQ;

    v4f acc = {};
    #pragma unroll 4
    for (int s = 0; s < SEQ; ++s) {
        float mm = (float)m[s];                                // uniform per block
        v4f v = __builtin_nontemporal_load(row + s * (FEAT / 4) + g);
        acc.x = fmaf(mm, v.x, acc.x);
        acc.y = fmaf(mm, v.y, acc.y);
        acc.z = fmaf(mm, v.z, acc.z);
        acc.w = fmaf(mm, v.w, acc.w);
    }
    v4h h;
    h[0] = (half_t)acc.x; h[1] = (half_t)acc.y;
    h[2] = (half_t)acc.z; h[3] = (half_t)acc.w;
    *(v4h*)(xh + (size_t)b * FEAT + g * 4) = h;
}

// ---------------------------------------------------------------------------
// Kernel 3: fused 2-layer MLP with tanh, f16 WMMA (v_wmma_f32_16x16x32_f16).
// Block = 4 waves (128 threads), computes one 16-row stripe of the batch.
// ---------------------------------------------------------------------------
// A-fragment ISA layout (16-bit A 16x32): lane L<16 holds row L,
// K = {kb..kb+7, kb+16..kb+23}; lane L>=16 holds row L-16 with +8 K offset.
// B is supplied column-per-lane with the identical K striping; since
// out = x @ W^T, B column j == W row j (contiguous), so both halves are
// straight b128 loads.
template <bool LAST>
__device__ __forceinline__ void mlp_layer(const half_t* __restrict__ Asrc,   // LDS [16][256]
                                          const half_t* __restrict__ Wh,     // global [256][256]
                                          const float*  __restrict__ bias,   // global [256]
                                          half_t* __restrict__ Hdst,         // LDS [16][256] (LAST=false)
                                          float*  __restrict__ outg,         // global (LAST=true)
                                          int wave, int lane, int r0) {
    const int mn    = lane & 15;       // A row / B-C-D column within tile
    const int khalf = lane >> 4;       // 0 or 1: which K-half this lane holds

    v8f acc[4] = {};
    #pragma unroll
    for (int ks = 0; ks < 8; ++ks) {
        const int kb = ks * 32 + khalf * 8;
        // A fragment: two b128 LDS loads
        v8h alo = *(const v8h*)(Asrc + mn * FEAT + kb);
        v8h ahi = *(const v8h*)(Asrc + mn * FEAT + kb + 16);
        v16h a;
        #pragma unroll
        for (int e = 0; e < 8; ++e) { a[e] = alo[e]; a[8 + e] = ahi[e]; }

        #pragma unroll
        for (int t = 0; t < 4; ++t) {
            const int j = (wave * 4 + t) * 16 + mn;            // output column
            v8h blo = *(const v8h*)(Wh + (size_t)j * FEAT + kb);
            v8h bhi = *(const v8h*)(Wh + (size_t)j * FEAT + kb + 16);
            v16h bb;
            #pragma unroll
            for (int e = 0; e < 8; ++e) { bb[e] = blo[e]; bb[8 + e] = bhi[e]; }

            acc[t] = __builtin_amdgcn_wmma_f32_16x16x32_f16(
                false, a, false, bb, (short)0, acc[t], false, false);
        }
    }

    // Epilogue: bias + tanh. C/D layout: lanes 0-15 -> M=v, N=lane;
    // lanes 16-31 -> M=v+8, N=lane-16.
    const int mbase = (lane >> 4) * 8;
    #pragma unroll
    for (int t = 0; t < 4; ++t) {
        const int j = (wave * 4 + t) * 16 + mn;
        const float bj = bias[j];
        #pragma unroll
        for (int v = 0; v < 8; ++v) {
            const int m = mbase + v;
            float val = tanhf(acc[t][v] + bj);
            if (LAST) {
                outg[(size_t)(r0 + m) * FEAT + j] = val;
            } else {
                Hdst[m * FEAT + j] = (half_t)val;
            }
        }
    }
}

__global__ __launch_bounds__(128) void mlp_kernel(const half_t* __restrict__ xh,
                                                  const half_t* __restrict__ W1h,
                                                  const float*  __restrict__ b1,
                                                  const half_t* __restrict__ W2h,
                                                  const float*  __restrict__ b2,
                                                  float* __restrict__ out) {
    __shared__ half_t Ain[16 * FEAT];   // 8 KB: layer-1 input rows
    __shared__ half_t Hid[16 * FEAT];   // 8 KB: layer-1 output rows

    const int r0   = blockIdx.x * 16;
    const int tid  = threadIdx.x;
    const int wave = tid >> 5;
    const int lane = tid & 31;

    // Stage 16 rows of pooled x into LDS (4096 halfs = 512 x v8h).
    {
        const v8h* src = (const v8h*)(xh + (size_t)r0 * FEAT);
        v8h* dst = (v8h*)Ain;
        #pragma unroll
        for (int i = 0; i < 4; ++i) dst[tid + i * 128] = src[tid + i * 128];
    }
    __syncthreads();

    mlp_layer<false>(Ain, W1h, b1, Hid, nullptr, wave, lane, r0);
    __syncthreads();
    mlp_layer<true >(Hid, W2h, b2, nullptr, out,  wave, lane, r0);
}

// ---------------------------------------------------------------------------
extern "C" void kernel_launch(void* const* d_in, const int* in_sizes, int n_in,
                              void* d_out, int out_size, void* d_ws, size_t ws_size,
                              hipStream_t stream) {
    const float*         te   = (const float*)d_in[0];
    const unsigned char* mask = (const unsigned char*)d_in[1];   // jnp.bool_: 1 byte/elt
    const float*         W1   = (const float*)d_in[2];
    const float*         b1   = (const float*)d_in[3];
    const float*         W2   = (const float*)d_in[4];
    const float*         b2   = (const float*)d_in[5];
    float* out = (float*)d_out;

    char* ws = (char*)d_ws;
    half_t* xh  = (half_t*)ws;                                        // 2 MiB
    half_t* W1h = (half_t*)(ws + (size_t)BATCH * FEAT * sizeof(half_t));
    half_t* W2h = W1h + FEAT * FEAT;                                  // +128 KiB each

    cvt_w_kernel<<<(FEAT * FEAT + 255) / 256, 256, 0, stream>>>(W1, W2, W1h, W2h);
    pool_kernel<<<BATCH, 64, 0, stream>>>(te, mask, xh);
    mlp_kernel<<<BATCH / 16, 128, 0, stream>>>(xh, W1h, b1, W2h, b2, out);
}
